// SequenceEncoder_120259085197
// MI455X (gfx1250) — matrix operand order repaired and checked by reading, not compiled, not run on hardware
//
#include <hip/hip_runtime.h>
#include <math.h>

#define BATCH  64
#define SEQL   512
#define DIM    1024
#define LATENT 512
#define TQ     128      // block tile: q rows
#define TK     64       // block tile: k rows
#define KC     32       // d-chunk per WMMA sweep
#define NCHUNK (DIM / KC)

typedef __attribute__((ext_vector_type(16))) __bf16 v16bf;
typedef __attribute__((ext_vector_type(8)))  float  v8f;
typedef __attribute__((ext_vector_type(2)))  float  v2f;
typedef __attribute__((ext_vector_type(4)))  int    v4i;

union FragBF { uint4 u[2]; v16bf v; };

__device__ __forceinline__ unsigned short f2bf(float f) {
    unsigned int u = __float_as_uint(f);
    u += 0x7FFFu + ((u >> 16) & 1u);            // round-to-nearest-even
    return (unsigned short)(u >> 16);
}

#if __has_builtin(__builtin_amdgcn_global_load_async_to_lds_b128) && \
    __has_builtin(__builtin_amdgcn_s_wait_asynccnt)
#define USE_ASYNC 1
#else
#define USE_ASYNC 0
#endif

#if USE_ASYNC
// 16-byte async copy global -> LDS (per-lane, tracked by ASYNCcnt).
__device__ __forceinline__ void async_cp16(const unsigned short* g, unsigned short* l) {
    __attribute__((address_space(1))) v4i* gp =
        (__attribute__((address_space(1))) v4i*)(unsigned long long)g;
    __attribute__((address_space(3))) v4i* lp =
        (__attribute__((address_space(3))) v4i*)(unsigned long long)(unsigned)(unsigned long long)l;
    __builtin_amdgcn_global_load_async_to_lds_b128(gp, lp, 0, 0);
}
#endif

// ---------------------------------------------------------------------------
// Kernel 0: gather + convert ctx rows to bf16 once.  ctxb[(b*SEQL+k)*DIM+d]
// ---------------------------------------------------------------------------
__global__ __launch_bounds__(256) void convert_kernel(
    const float* __restrict__ seqs, const int* __restrict__ sse,
    unsigned short* __restrict__ ctxb)
{
    const size_t i8 = ((size_t)blockIdx.x * 256 + threadIdx.x) * 8;
    const int row = (int)(i8 / DIM);            // b*SEQL + k
    const int col = (int)(i8 % DIM);
    const int b = row >> 9, k = row & (SEQL - 1);
    const int start = sse[2 * b];
    const float4 f0 = *(const float4*)(seqs + (size_t)(start + k) * DIM + col);
    const float4 f1 = *(const float4*)(seqs + (size_t)(start + k) * DIM + col + 4);
    uint4 p;
    p.x = (unsigned)f2bf(f0.x) | ((unsigned)f2bf(f0.y) << 16);
    p.y = (unsigned)f2bf(f0.z) | ((unsigned)f2bf(f0.w) << 16);
    p.z = (unsigned)f2bf(f1.x) | ((unsigned)f2bf(f1.y) << 16);
    p.w = (unsigned)f2bf(f1.z) | ((unsigned)f2bf(f1.w) << 16);
    *(uint4*)(ctxb + i8) = p;
}

// ---------------------------------------------------------------------------
// Kernel 1a (fast path): scores from pre-converted bf16 ctx.
// 128x64 tile/block, 8 waves, each a 32x32 quadrant (2x2 WMMA 16x16x32 bf16).
// Triple-buffered async global->LDS staging: chunk ci+2 issued while chunk ci
// computes; s_wait_asynccnt(3) = "everything but the newest chunk landed".
// ---------------------------------------------------------------------------
__global__ __launch_bounds__(256) void scores_bf16_kernel(
    const unsigned short* __restrict__ ctxb, const int* __restrict__ sse,
    float* __restrict__ S)
{
    __shared__ __align__(16) unsigned short sA[3][TQ * 40];  // padded stride 40
    __shared__ __align__(16) unsigned short sB[3][TK * 40];

    const int kt = blockIdx.x, qt = blockIdx.y, b = blockIdx.z;
    const int len   = sse[2 * b + 1] - sse[2 * b];
    const int tid   = threadIdx.x;
    const int lane  = tid & 31;
    const int wave  = tid >> 5;
    const int wq    = (wave >> 1) * 32;
    const int wk    = (wave & 1) * 32;
    const int qBase = qt * TQ, kBase = kt * TK;
    const size_t rowA0 = (size_t)b * SEQL + qBase;   // ctxb row base (A)
    const size_t rowB0 = (size_t)b * SEQL + kBase;   // ctxb row base (B)

    // Staging units: 16B = 8 halfs. A: 128 rows x 4 units, B: 64 x 4.
    // 768 units / 256 threads = 3 per thread: i=0,1 -> A, i=2 -> B.
#if USE_ASYNC
    auto issue = [&](int buf, int d0) {
        #pragma unroll
        for (int i = 0; i < 2; ++i) {
            const int u = tid + i * 256;
            const int row = u >> 2, seg = (u & 3) * 8;
            async_cp16(ctxb + (rowA0 + row) * DIM + d0 + seg,
                       &sA[buf][row * 40 + seg]);
        }
        {
            const int row = tid >> 2, seg = (tid & 3) * 8;
            async_cp16(ctxb + (rowB0 + row) * DIM + d0 + seg,
                       &sB[buf][row * 40 + seg]);
        }
    };
#else
    uint4 st[3];
    auto loadRegs = [&](int d0) {
        #pragma unroll
        for (int i = 0; i < 2; ++i) {
            const int u = tid + i * 256;
            const int row = u >> 2, seg = (u & 3) * 8;
            st[i] = *(const uint4*)(ctxb + (rowA0 + row) * DIM + d0 + seg);
        }
        st[2] = *(const uint4*)(ctxb + (rowB0 + (tid >> 2)) * DIM + d0 + (tid & 3) * 8);
    };
    auto storeLds = [&](int buf) {
        #pragma unroll
        for (int i = 0; i < 2; ++i) {
            const int u = tid + i * 256;
            *(uint4*)&sA[buf][(u >> 2) * 40 + (u & 3) * 8] = st[i];
        }
        *(uint4*)&sB[buf][(tid >> 2) * 40 + (tid & 3) * 8] = st[2];
    };
#endif

    const v8f vzero = {0.f, 0.f, 0.f, 0.f, 0.f, 0.f, 0.f, 0.f};
    v8f c[2][2];
    c[0][0] = vzero; c[0][1] = vzero; c[1][0] = vzero; c[1][1] = vzero;

    const int arow = wq + (lane & 15);
    const int akb  = (lane < 16) ? 0 : 8;
    const int bcol = wk + (lane & 15);
    const int bkb  = (lane < 16) ? 0 : 8;

#if USE_ASYNC
    issue(0, 0);
    issue(1, KC);
    for (int ci = 0; ci < NCHUNK; ++ci) {
        const int cur = ci % 3;
        // In-order completion: <=3 outstanding means chunk ci has fully landed
        // (only chunk ci+1's 3 copies may remain in flight).
        if (ci < NCHUNK - 1) __builtin_amdgcn_s_wait_asynccnt(3);
        else                 __builtin_amdgcn_s_wait_asynccnt(0);
        __syncthreads();                          // all waves' writes visible
        if (ci + 2 < NCHUNK) issue((ci + 2) % 3, (ci + 2) * KC);
#else
    loadRegs(0);
    storeLds(0);
    for (int ci = 0; ci < NCHUNK; ++ci) {
        const int cur = ci & 1;
        __syncthreads();
        if (ci + 1 < NCHUNK) loadRegs((ci + 1) * KC);
#endif
        FragBF a[2], bb[2];
        #pragma unroll
        for (int mi = 0; mi < 2; ++mi) {
            const unsigned short* pa = &sA[cur][(arow + mi * 16) * 40 + akb];
            a[mi].u[0] = *(const uint4*)pa;
            a[mi].u[1] = *(const uint4*)(pa + 16);
        }
        #pragma unroll
        for (int ni = 0; ni < 2; ++ni) {
            const unsigned short* pb = &sB[cur][(bcol + ni * 16) * 40 + bkb];
            bb[ni].u[0] = *(const uint4*)pb;
            bb[ni].u[1] = *(const uint4*)(pb + 16);
        }
        #pragma unroll
        for (int mi = 0; mi < 2; ++mi)
            #pragma unroll
            for (int ni = 0; ni < 2; ++ni)
                c[mi][ni] = __builtin_amdgcn_wmma_f32_16x16x32_bf16(
                    false, a[mi].v, false, bb[ni].v, (short)0, c[mi][ni], false, false);
#if !USE_ASYNC
        if (ci + 1 < NCHUNK) storeLds((ci + 1) & 1);
#endif
    }

    const float scale = 0.03125f;  // 1/sqrt(1024)
    #pragma unroll
    for (int mi = 0; mi < 2; ++mi) {
        #pragma unroll
        for (int ni = 0; ni < 2; ++ni) {
            const int kg = kBase + wk + ni * 16 + (lane & 15);
            const bool masked = (kg >= len);
            #pragma unroll
            for (int i = 0; i < 8; ++i) {
                const int qg = qBase + wq + mi * 16 + i + ((lane < 16) ? 0 : 8);
                float v = c[mi][ni][i] * scale;
                if (masked) v = -INFINITY;
                S[((size_t)b * SEQL + qg) * SEQL + kg] = v;
            }
        }
    }
}

// ---------------------------------------------------------------------------
// Kernel 1b (fallback if ws too small): on-the-fly f32->bf16 conversion.
// ---------------------------------------------------------------------------
__global__ __launch_bounds__(256) void scores_f32_kernel(
    const float* __restrict__ seqs, const int* __restrict__ sse,
    float* __restrict__ S)
{
    __shared__ __align__(16) unsigned short sA[2][TQ * 40];
    __shared__ __align__(16) unsigned short sB[2][TK * 40];

    const int kt = blockIdx.x, qt = blockIdx.y, b = blockIdx.z;
    const int start = sse[2 * b];
    const int len   = sse[2 * b + 1] - start;
    const int tid   = threadIdx.x;
    const int lane  = tid & 31;
    const int wave  = tid >> 5;
    const int wq    = (wave >> 1) * 32;
    const int wk    = (wave & 1) * 32;
    const int qBase = qt * TQ, kBase = kt * TK;

    float4 st[6];
    auto loadRegs = [&](int d0) {
        #pragma unroll
        for (int i = 0; i < 4; ++i) {
            const int u = tid + i * 256;
            st[i] = *(const float4*)(seqs + (size_t)(start + qBase + (u >> 3)) * DIM + d0 + (u & 7) * 4);
        }
        #pragma unroll
        for (int i = 0; i < 2; ++i) {
            const int u = tid + i * 256;
            st[4 + i] = *(const float4*)(seqs + (size_t)(start + kBase + (u >> 3)) * DIM + d0 + (u & 7) * 4);
        }
    };
    auto storeLds = [&](int buf) {
        #pragma unroll
        for (int i = 0; i < 4; ++i) {
            const int u = tid + i * 256;
            uint2 p;
            p.x = (unsigned)f2bf(st[i].x) | ((unsigned)f2bf(st[i].y) << 16);
            p.y = (unsigned)f2bf(st[i].z) | ((unsigned)f2bf(st[i].w) << 16);
            *(uint2*)&sA[buf][(u >> 3) * 40 + (u & 7) * 4] = p;
        }
        #pragma unroll
        for (int i = 0; i < 2; ++i) {
            const int u = tid + i * 256;
            uint2 p;
            p.x = (unsigned)f2bf(st[4 + i].x) | ((unsigned)f2bf(st[4 + i].y) << 16);
            p.y = (unsigned)f2bf(st[4 + i].z) | ((unsigned)f2bf(st[4 + i].w) << 16);
            *(uint2*)&sB[buf][(u >> 3) * 40 + (u & 7) * 4] = p;
        }
    };

    const v8f vzero = {0.f, 0.f, 0.f, 0.f, 0.f, 0.f, 0.f, 0.f};
    v8f c[2][2];
    c[0][0] = vzero; c[0][1] = vzero; c[1][0] = vzero; c[1][1] = vzero;

    loadRegs(0);
    storeLds(0);
    int cur = 0;
    const int arow = wq + (lane & 15);
    const int akb  = (lane < 16) ? 0 : 8;
    const int bcol = wk + (lane & 15);
    const int bkb  = (lane < 16) ? 0 : 8;

    for (int ci = 0; ci < NCHUNK; ++ci) {
        __syncthreads();
        if (ci + 1 < NCHUNK) loadRegs((ci + 1) * KC);
        FragBF a[2], bb[2];
        #pragma unroll
        for (int mi = 0; mi < 2; ++mi) {
            const unsigned short* pa = &sA[cur][(arow + mi * 16) * 40 + akb];
            a[mi].u[0] = *(const uint4*)pa;
            a[mi].u[1] = *(const uint4*)(pa + 16);
        }
        #pragma unroll
        for (int ni = 0; ni < 2; ++ni) {
            const unsigned short* pb = &sB[cur][(bcol + ni * 16) * 40 + bkb];
            bb[ni].u[0] = *(const uint4*)pb;
            bb[ni].u[1] = *(const uint4*)(pb + 16);
        }
        #pragma unroll
        for (int mi = 0; mi < 2; ++mi)
            #pragma unroll
            for (int ni = 0; ni < 2; ++ni)
                c[mi][ni] = __builtin_amdgcn_wmma_f32_16x16x32_bf16(
                    false, a[mi].v, false, bb[ni].v, (short)0, c[mi][ni], false, false);
        if (ci + 1 < NCHUNK) storeLds(cur ^ 1);
        cur ^= 1;
    }

    const float scale = 0.03125f;
    #pragma unroll
    for (int mi = 0; mi < 2; ++mi) {
        #pragma unroll
        for (int ni = 0; ni < 2; ++ni) {
            const int kg = kBase + wk + ni * 16 + (lane & 15);
            const bool masked = (kg >= len);
            #pragma unroll
            for (int i = 0; i < 8; ++i) {
                const int qg = qBase + wq + mi * 16 + i + ((lane < 16) ? 0 : 8);
                float v = c[mi][ni][i] * scale;
                if (masked) v = -INFINITY;
                S[((size_t)b * SEQL + qg) * SEQL + kg] = v;
            }
        }
    }
}

// ---------------------------------------------------------------------------
// Kernel 2: per-row softmax stats (max, 1/sum-exp). One wave32 per row.
// ---------------------------------------------------------------------------
__global__ __launch_bounds__(256) void stats_kernel(
    const float* __restrict__ S, float* __restrict__ mrow, float* __restrict__ zinv)
{
    const int r    = blockIdx.x * 8 + (threadIdx.x >> 5);
    const int lane = threadIdx.x & 31;
    const float* row = S + (size_t)r * SEQL;
    float m = -INFINITY;
    #pragma unroll
    for (int i = 0; i < 16; ++i) m = fmaxf(m, row[lane + i * 32]);
    #pragma unroll
    for (int off = 16; off > 0; off >>= 1) m = fmaxf(m, __shfl_xor(m, off, 32));
    float z = 0.f;
    #pragma unroll
    for (int i = 0; i < 16; ++i) z += __expf(row[lane + i * 32] - m);
    #pragma unroll
    for (int off = 16; off > 0; off >>= 1) z += __shfl_xor(z, off, 32);
    if (lane == 0) { mrow[r] = m; zinv[r] = 1.0f / z; }
}

// ---------------------------------------------------------------------------
// Kernel 3: colw[b,k] = sum_{q<len} exp(S[q,k]-m_q)/Z_q  (coalesced over k).
// ---------------------------------------------------------------------------
__global__ __launch_bounds__(512) void colsum_kernel(
    const float* __restrict__ S, const float* __restrict__ mrow,
    const float* __restrict__ zinv, const int* __restrict__ sse,
    float* __restrict__ colw)
{
    const int b = blockIdx.x;
    const int k = threadIdx.x;
    const int len = sse[2 * b + 1] - sse[2 * b];
    const float* Sb = S + (size_t)b * SEQL * SEQL;
    const float* mb = mrow + b * SEQL;
    const float* zb = zinv + b * SEQL;
    float acc = 0.f;
    for (int q = 0; q < len; ++q) {
        __builtin_prefetch(Sb + (size_t)(q + 16) * SEQL + k, 0, 1);
        acc += __expf(Sb[(size_t)q * SEQL + k] - mb[q]) * zb[q];
    }
    colw[b * SEQL + k] = acc;
}

// ---------------------------------------------------------------------------
// Kernel 4: pooled[b,d] = sum_k colw[b,k] * ctx[b,k,d]   (full f32 precision)
// ---------------------------------------------------------------------------
__global__ __launch_bounds__(256) void pooled_kernel(
    const float* __restrict__ seqs, const int* __restrict__ sse,
    const float* __restrict__ colw, float* __restrict__ pooled)
{
    const int b = blockIdx.y;
    const int d = blockIdx.x * 256 + threadIdx.x;
    const int start = sse[2 * b];
    const int len   = sse[2 * b + 1] - start;
    const float* cw = colw + b * SEQL;
    float acc = 0.f;
    for (int k = 0; k < len; ++k) {
        __builtin_prefetch(seqs + (size_t)(start + k + 16) * DIM + d, 0, 1);
        acc += cw[k] * seqs[(size_t)(start + k) * DIM + d];
    }
    pooled[(size_t)b * DIM + d] = acc;
}

// ---------------------------------------------------------------------------
// Kernel 5: mean / log_var = pooled @ W + bias via f32 WMMA 16x16x4.
// ---------------------------------------------------------------------------
__global__ __launch_bounds__(128) void head_kernel(
    const float* __restrict__ pooled,
    const float* __restrict__ Wm, const float* __restrict__ bm,
    const float* __restrict__ Wl, const float* __restrict__ bl,
    float* __restrict__ out)
{
    const int which = blockIdx.y;
    const float* W    = which ? Wl : Wm;
    const float* bias = which ? bl : bm;
    const int lane = threadIdx.x & 31;
    const int wave = threadIdx.x >> 5;
    const int m0   = (blockIdx.x & 3) * 16;
    const int n0   = (blockIdx.x >> 2) * 64 + wave * 16;
    const int col  = lane & 15;
    const int ka   = (lane < 16) ? 0 : 2;
    const int M    = m0 + col;

    v8f c = {0.f, 0.f, 0.f, 0.f, 0.f, 0.f, 0.f, 0.f};
    for (int k = 0; k < DIM; k += 4) {
        const float2 af = *(const float2*)(pooled + (size_t)M * DIM + k + ka);
        v2f av; av.x = af.x; av.y = af.y;
        v2f bv;
        bv.x = W[(size_t)(k + ka)     * LATENT + n0 + col];
        bv.y = W[(size_t)(k + ka + 1) * LATENT + n0 + col];
        c = __builtin_amdgcn_wmma_f32_16x16x4_f32(
            false, av, false, bv, (short)0, c, false, false);
    }
    const float bs = bias[n0 + col];
    #pragma unroll
    for (int i = 0; i < 8; ++i) {
        const int row = m0 + i + ((lane < 16) ? 0 : 8);
        out[(size_t)which * BATCH * LATENT + (size_t)row * LATENT + n0 + col] = c[i] + bs;
    }
}

// ---------------------------------------------------------------------------
extern "C" void kernel_launch(void* const* d_in, const int* in_sizes, int n_in,
                              void* d_out, int out_size, void* d_ws, size_t ws_size,
                              hipStream_t stream)
{
    const float* seqs = (const float*)d_in[0];
    const float* Wm   = (const float*)d_in[1];
    const float* bm   = (const float*)d_in[2];
    const float* Wl   = (const float*)d_in[3];
    const float* bl   = (const float*)d_in[4];
    const int*   sse  = (const int*)d_in[5];
    float* out = (float*)d_out;

    float* ws     = (float*)d_ws;
    float* S      = ws;                                   // 64*512*512
    float* mrow   = S    + (size_t)BATCH * SEQL * SEQL;
    float* zinv   = mrow + (size_t)BATCH * SEQL;
    float* colw   = zinv + (size_t)BATCH * SEQL;
    float* pooled = colw + (size_t)BATCH * SEQL;
    unsigned short* ctxb = (unsigned short*)(pooled + (size_t)BATCH * DIM);

    const size_t baseBytes = (size_t)((char*)ctxb - (char*)d_ws);
    const size_t ctxbBytes = (size_t)BATCH * SEQL * DIM * sizeof(unsigned short);
    const bool pre = ws_size >= baseBytes + ctxbBytes;

    if (pre) {
        convert_kernel<<<(BATCH * SEQL * (DIM / 8)) / 256, 256, 0, stream>>>(seqs, sse, ctxb);
        scores_bf16_kernel<<<dim3(SEQL / TK, SEQL / TQ, BATCH), 256, 0, stream>>>(ctxb, sse, S);
    } else {
        scores_f32_kernel<<<dim3(SEQL / TK, SEQL / TQ, BATCH), 256, 0, stream>>>(seqs, sse, S);
    }
    stats_kernel <<<(BATCH * SEQL) / 8, 256, 0, stream>>>(S, mrow, zinv);
    colsum_kernel<<<BATCH, SEQL, 0, stream>>>(S, mrow, zinv, sse, colw);
    pooled_kernel<<<dim3(DIM / 256, BATCH), 256, 0, stream>>>(seqs, sse, colw, pooled);
    head_kernel  <<<dim3((LATENT / 64) * 4, 2), 128, 0, stream>>>(pooled, Wm, bm, Wl, bl, out);
}
